// SelfAttention_9577777070283
// MI455X (gfx1250) — matrix operand order repaired and checked by reading, not compiled, not run on hardware
//
#include <hip/hip_runtime.h>
#include <hip/hip_bf16.h>
#include <stdint.h>

typedef __attribute__((ext_vector_type(16))) __bf16 v16bf;
typedef __attribute__((ext_vector_type(4)))  __bf16 v4bf;
typedef __attribute__((ext_vector_type(8)))  float  v8f;

#define HID 4096
#define NH 32
#define NKV 8
#define HD 128
#define PAGE_SZ 16
#define MAXP 2048
#define B_ 4
#define QLEN 512
#define PREV 1016
#define PREV_PAGES 64
#define LAST_LEN 8
#define NEW_PAGES 32
#define TOT_PAGES 96
#define KV_LEN 1536
#define TOT 1528
#define NEWPAGE_BASE 256            /* B_*PREV_PAGES */
#define PAGE_STRIDE (PAGE_SZ * NKV * HD)   /* floats per (page, k-or-v) plane */
#define SCALE 0.08838834764831845f  /* 1/sqrt(128) */
#define LOG_THETA 9.210340371976184f

// ---------------------------------------------------------------------------
// CDNA5 async copy: global -> LDS, 16B per lane, tracked by ASYNCcnt.
// lds = wave-relative LDS byte offset (low 32 bits of generic LDS address).
// ---------------------------------------------------------------------------
__device__ __forceinline__ void async_b128(uint32_t lds, const __bf16* g) {
  asm volatile("global_load_async_to_lds_b128 %0, %1, off"
               :: "v"(lds), "v"(g) : "memory");
}
__device__ __forceinline__ void wait_async0() {
  asm volatile("s_wait_asynccnt 0x0" ::: "memory");
}

// ---------------------------------------------------------------------------
// WMMA fragment gathers from LDS, per CDNA5 ISA VGPR layouts (wave32).
// A (16x32 bf16): lane L holds row L&15; VGPR q holds K = 2q + (q>=4?8:0) + 8*(L>>4), +1
// B (32x16 bf16): lane L holds col L&15; K = e + (e>=8?8:0) + 8*(L>>4)
// C/D (16x16 f32): lane L, VGPR i -> row i + 8*(L>>4), col L&15
// ---------------------------------------------------------------------------
__device__ __forceinline__ v16bf frag_a_lds(const __bf16* tile, int ldt, int lane) {
  const __bf16* row = tile + (size_t)(lane & 15) * ldt;
  int hi = (lane >> 4) << 3;
  v16bf f;
#pragma unroll
  for (int e = 0; e < 16; ++e) {
    int q = e >> 1;
    int k = (q << 1) + ((q & 4) ? 8 : 0) + hi + (e & 1);
    f[e] = row[k];
  }
  return f;
}

// element (k, n): value = tile[n*sn + k*sk]
__device__ __forceinline__ v16bf frag_b_lds(const __bf16* tile, int sk, int sn, int lane) {
  const __bf16* base = tile + (size_t)(lane & 15) * sn;
  int kb = (lane >> 4) << 3;
  v16bf f;
#pragma unroll
  for (int e = 0; e < 16; ++e) f[e] = base[(size_t)(kb + e + ((e & 8) ? 8 : 0)) * sk];
  return f;
}

// ---------------------------------------------------------------------------
// fp32 -> bf16 conversion pass (one-time; weights are reused by many blocks).
// ---------------------------------------------------------------------------
__global__ __launch_bounds__(256) void cvt_f32_bf16(const float* __restrict__ src,
                                                    __bf16* __restrict__ dst, int n4) {
  int i = blockIdx.x * 256 + threadIdx.x;
  if (i < n4) {
    float4 v = ((const float4*)src)[i];
    v4bf o;
    o[0] = (__bf16)v.x; o[1] = (__bf16)v.y; o[2] = (__bf16)v.z; o[3] = (__bf16)v.w;
    ((v4bf*)dst)[i] = o;
  }
}

// ---------------------------------------------------------------------------
// GEMM: C[M,N] = A[M,K] * B[N,K]^T, bf16 operands, fp32 accumulate.
// 128x128 block tile, 8 waves x (32x64), K-step 32.
// Tiles staged with global_load_async_to_lds_b128, double buffered (ASYNCcnt).
// M%128==0, N%128==0, K%32==0.
// ---------------------------------------------------------------------------
#define LDT 40
#define TILE_E (128 * LDT)

__global__ __launch_bounds__(256) void gemm_bf16_async(
    const __bf16* __restrict__ A, const __bf16* __restrict__ B,
    float* __restrict__ C, int M, int N, int K) {
  __shared__ __align__(16) __bf16 As[2 * TILE_E];
  __shared__ __align__(16) __bf16 Bs[2 * TILE_E];
  int tid = threadIdx.x, lane = tid & 31, w = tid >> 5;
  int bm = blockIdx.y * 128, bn = blockIdx.x * 128;
  int m_off = (w >> 1) * 32, n_off = (w & 1) * 64;

  uint32_t asb = (uint32_t)(uintptr_t)&As[0];
  uint32_t bsb = (uint32_t)(uintptr_t)&Bs[0];

  v8f acc[2][4];
#pragma unroll
  for (int i = 0; i < 2; ++i)
#pragma unroll
    for (int j = 0; j < 4; ++j) acc[i][j] = v8f{0.f,0.f,0.f,0.f,0.f,0.f,0.f,0.f};

  // each thread owns 2 16-byte segments of each tile: 512 segs = 128 rows x 4
  auto issue = [&](int buf, int k0) {
#pragma unroll
    for (int q = 0; q < 2; ++q) {
      int idx = tid * 2 + q;
      int r = idx >> 2, sg = (idx & 3) << 3;          // sg: bf16 offset in row
      uint32_t loff = (uint32_t)((buf * TILE_E + r * LDT + sg) * 2);
      async_b128(asb + loff, A + (size_t)(bm + r) * K + k0 + sg);
      async_b128(bsb + loff, B + (size_t)(bn + r) * K + k0 + sg);
    }
  };

  issue(0, 0);
  int buf = 0;
  for (int k0 = 0; k0 < K; k0 += 32) {
    wait_async0();          // this wave's fills for `buf` have landed in LDS
    __syncthreads();        // everyone's fills visible
    const __bf16* at = As + buf * TILE_E;
    const __bf16* bt = Bs + buf * TILE_E;
    v16bf am[2], bnf[4];
#pragma unroll
    for (int i = 0; i < 2; ++i)
      am[i] = frag_a_lds(at + (m_off + 16 * i) * LDT, LDT, lane);
#pragma unroll
    for (int j = 0; j < 4; ++j)
      bnf[j] = frag_b_lds(bt + (n_off + 16 * j) * LDT, 1, LDT, lane);
#pragma unroll
    for (int i = 0; i < 2; ++i)
#pragma unroll
      for (int j = 0; j < 4; ++j)
        acc[i][j] = __builtin_amdgcn_wmma_f32_16x16x32_bf16(
            false, am[i], false, bnf[j], (short)0, acc[i][j], false, false);
    __syncthreads();        // all reads of `buf^1`'s target region done
    if (k0 + 32 < K) issue(buf ^ 1, k0 + 32);
    buf ^= 1;
  }

  int half = lane >> 4, col = lane & 15;
#pragma unroll
  for (int i = 0; i < 2; ++i)
#pragma unroll
    for (int j = 0; j < 4; ++j)
#pragma unroll
      for (int r = 0; r < 8; ++r) {
        int row = bm + m_off + 16 * i + r + half * 8;
        int cc  = bn + n_off + 16 * j + col;
        C[(size_t)row * N + cc] = acc[i][j][r];
      }
}

// ---------------------------------------------------------------------------
// KV append into the cache copy + index outputs.
// ---------------------------------------------------------------------------
__global__ __launch_bounds__(256) void kv_append_kernel(
    const float* __restrict__ Kw, const float* __restrict__ Vw,
    const int* __restrict__ kv_indices, float* __restrict__ cache,
    float* __restrict__ idx_out) {
  int idx = blockIdx.x * 256 + threadIdx.x;   // 0 .. 2097151
  if (idx < 5) {
    idx_out[idx] = (float)(idx * TOT_PAGES);
  } else if (idx < 389) {
    int j = idx - 5, bb = j / TOT_PAGES, jj = j % TOT_PAGES;
    int v = (jj < PREV_PAGES) ? kv_indices[bb * PREV_PAGES + jj]
                              : NEWPAGE_BASE + bb * NEW_PAGES + (jj - PREV_PAGES);
    idx_out[idx] = (float)v;
  } else if (idx < 393) {
    idx_out[idx] = (float)((LAST_LEN + QLEN - 1) % PAGE_SZ + 1);
  }
  int d = idx & 127, h = (idx >> 7) & 7, t = (idx >> 10) & 511, b = idx >> 19;
  int pos = LAST_LEN + t, p = pos >> 4, slot = pos & 15;
  int page = (p == 0) ? kv_indices[b * PREV_PAGES + PREV_PAGES - 1]
                      : NEWPAGE_BASE + b * NEW_PAGES + (p - 1);
  size_t base = (size_t)page * 2 * PAGE_STRIDE + (size_t)slot * (NKV * HD) + h * HD + d;
  size_t src  = ((size_t)(b * QLEN + t) * NKV + h) * HD + d;
  cache[base] = Kw[src];
  cache[base + PAGE_STRIDE] = Vw[src];
}

// ---------------------------------------------------------------------------
// Flash attention with RoPE, GQA (GRP=4), causal mask, paged KV.
// Block = 4 waves; wave handles a 16-row q tile; block covers 64 q rows of one
// (b, h). KV staged in 32-position chunks shared by all 4 waves.
// Output stored directly as bf16 for the final async GEMM.
// ---------------------------------------------------------------------------
#define AK_LD 136
#define AP_LD 40

__global__ __launch_bounds__(128) void attn_kernel(
    const float* __restrict__ Qw, const float* __restrict__ cache,
    const int* __restrict__ kv_indices, __bf16* __restrict__ Ow) {
  __shared__ __bf16 Kt[32 * AK_LD];
  __shared__ __bf16 Vt[32 * AK_LD];
  __shared__ __bf16 Qt[4 * 16 * AK_LD];
  __shared__ __bf16 Pt[4 * 16 * AP_LD];

  int tid = threadIdx.x, lane = tid & 31, w = tid >> 5;
  int qb0 = blockIdx.x * 64;
  int h = blockIdx.y, b = blockIdx.z, kvh = h >> 2;
  int half = lane >> 4, col = lane & 15;

  // ---- stage q tile: RoPE + fold in 1/sqrt(HD), fp32 -> bf16 ----
  __bf16* qtw = Qt + w * 16 * AK_LD;
  for (int j = 0; j < 64; ++j) {
    int li = lane + 32 * j;                // 16 rows * 128 dims
    int r = li >> 7, d = li & 127;
    int trow = qb0 + w * 16 + r;
    float pos = (float)(PREV + trow);
    int j0 = d & 63;
    float inv = __expf(-(float)j0 * (2.0f / HD) * LOG_THETA);
    float sn, cs; __sincosf(pos * inv, &sn, &cs);
    const float* qrow = Qw + (size_t)(b * QLEN + trow) * HID + h * HD;
    float x = qrow[d];
    float y = (d < 64) ? qrow[d + 64] : qrow[d - 64];
    float v = (d < 64) ? (x * cs - y * sn) : (x * cs + y * sn);
    qtw[r * AK_LD + d] = (__bf16)(v * SCALE);
  }
  __syncthreads();

  v16bf qf[4];
#pragma unroll
  for (int ks = 0; ks < 4; ++ks) qf[ks] = frag_a_lds(qtw + ks * 32, AK_LD, lane);

  v8f accO[8];
#pragma unroll
  for (int t = 0; t < 8; ++t) accO[t] = v8f{0.f,0.f,0.f,0.f,0.f,0.f,0.f,0.f};
  float mrun[8], lrun[8];
#pragma unroll
  for (int i = 0; i < 8; ++i) { mrun[i] = -1e30f; lrun[i] = 0.f; }

  int nchunk = (PREV + qb0 + 64 + 31) >> 5;
  for (int ck = 0; ck < nchunk; ++ck) {
    __syncthreads();
    // ---- stage K (RoPE'd) and V chunk from paged cache ----
    for (int j = 0; j < 32; ++j) {
      int li = tid + 128 * j;              // 32 positions * 128 dims
      int kl = li >> 7, d = li & 127;
      int pos = ck * 32 + kl;
      int p = pos >> 4;
      int page = (p < PREV_PAGES) ? kv_indices[b * PREV_PAGES + p]
                                  : NEWPAGE_BASE + b * NEW_PAGES + (p - PREV_PAGES);
      const float* kp = cache + (size_t)page * 2 * PAGE_STRIDE
                        + (size_t)(pos & 15) * (NKV * HD) + kvh * HD;
      float x = kp[d];
      float y = (d < 64) ? kp[d + 64] : kp[d - 64];
      int j0 = d & 63;
      float inv = __expf(-(float)j0 * (2.0f / HD) * LOG_THETA);
      float sn, cs; __sincosf((float)pos * inv, &sn, &cs);
      float kv = (d < 64) ? (x * cs - y * sn) : (x * cs + y * sn);
      Kt[kl * AK_LD + d] = (__bf16)kv;
      Vt[kl * AK_LD + d] = (__bf16)kp[PAGE_STRIDE + d];
    }
    __syncthreads();

    // ---- scores: S(16x32) = q(16x128) . Kr^T(128x32) ----
    v8f s0 = v8f{0.f,0.f,0.f,0.f,0.f,0.f,0.f,0.f};
    v8f s1 = v8f{0.f,0.f,0.f,0.f,0.f,0.f,0.f,0.f};
#pragma unroll
    for (int ks = 0; ks < 4; ++ks) {
      v16bf kb0 = frag_b_lds(Kt + ks * 32, 1, AK_LD, lane);
      v16bf kb1 = frag_b_lds(Kt + 16 * AK_LD + ks * 32, 1, AK_LD, lane);
      s0 = __builtin_amdgcn_wmma_f32_16x16x32_bf16(false, qf[ks], false, kb0, (short)0, s0, false, false);
      s1 = __builtin_amdgcn_wmma_f32_16x16x32_bf16(false, qf[ks], false, kb1, (short)0, s1, false, false);
    }

    // ---- mask + online softmax (row reductions across 16-lane half) ----
    float sfv[8];
#pragma unroll
    for (int i = 0; i < 8; ++i) {
      int qrow = PREV + qb0 + w * 16 + i + half * 8;
      int p0 = ck * 32 + col, p1 = p0 + 16;
      float v0 = (p0 <= qrow && p0 < TOT) ? s0[i] : -1e30f;
      float v1 = (p1 <= qrow && p1 < TOT) ? s1[i] : -1e30f;
      float mx = fmaxf(v0, v1);
      mx = fmaxf(mx, __shfl_xor(mx, 1, 32));
      mx = fmaxf(mx, __shfl_xor(mx, 2, 32));
      mx = fmaxf(mx, __shfl_xor(mx, 4, 32));
      mx = fmaxf(mx, __shfl_xor(mx, 8, 32));
      float mnew = fmaxf(mrun[i], mx);
      float sf = __expf(mrun[i] - mnew);
      mrun[i] = mnew;
      float p0e = __expf(v0 - mnew), p1e = __expf(v1 - mnew);
      float rs = p0e + p1e;
      rs += __shfl_xor(rs, 1, 32);
      rs += __shfl_xor(rs, 2, 32);
      rs += __shfl_xor(rs, 4, 32);
      rs += __shfl_xor(rs, 8, 32);
      lrun[i] = lrun[i] * sf + rs;
      sfv[i] = sf;
      __bf16* pr = Pt + (w * 16 + i + half * 8) * AP_LD;
      pr[col] = (__bf16)p0e;
      pr[col + 16] = (__bf16)p1e;
    }
#pragma unroll
    for (int t = 0; t < 8; ++t)
#pragma unroll
      for (int i = 0; i < 8; ++i) accO[t][i] *= sfv[i];
    __syncthreads();

    // ---- O += P(16x32) . V(32x128) ----
    v16bf pf = frag_a_lds(Pt + w * 16 * AP_LD, AP_LD, lane);
#pragma unroll
    for (int t = 0; t < 8; ++t) {
      v16bf vb = frag_b_lds(Vt + t * 16, AK_LD, 1, lane);
      accO[t] = __builtin_amdgcn_wmma_f32_16x16x32_bf16(false, pf, false, vb, (short)0, accO[t], false, false);
    }
  }

  // ---- normalize and store O[b, q, h, d] as bf16 [B*QLEN, HID] ----
  float linv[8];
#pragma unroll
  for (int i = 0; i < 8; ++i) linv[i] = 1.0f / lrun[i];
#pragma unroll
  for (int t = 0; t < 8; ++t)
#pragma unroll
    for (int i = 0; i < 8; ++i) {
      int m = i + half * 8;
      int row = b * QLEN + qb0 + w * 16 + m;
      int dcol = h * HD + t * 16 + col;
      Ow[(size_t)row * HID + dcol] = (__bf16)(accO[t][i] * linv[i]);
    }
}

// ---------------------------------------------------------------------------
extern "C" void kernel_launch(void* const* d_in, const int* in_sizes, int n_in,
                              void* d_out, int out_size, void* d_ws, size_t ws_size,
                              hipStream_t stream) {
  (void)in_sizes; (void)n_in; (void)out_size; (void)ws_size;
  const float* hidden   = (const float*)d_in[0];
  const float* cache_in = (const float*)d_in[1];
  const float* Wq = (const float*)d_in[2];
  const float* Wk = (const float*)d_in[3];
  const float* Wv = (const float*)d_in[4];
  const float* Wo = (const float*)d_in[5];
  const int* kv_indices = (const int*)d_in[8];

  const size_t M = (size_t)B_ * QLEN;          // 2048
  const size_t KVD = (size_t)NKV * HD;         // 1024
  float* out       = (float*)d_out;            // [2048*4096]
  float* cache_out = out + M * HID;            // [MAXP*2*16*8*128]
  float* idx_out   = cache_out + (size_t)MAXP * 2 * PAGE_STRIDE;

  char* p = (char*)d_ws;
  __bf16* hid_bf = (__bf16*)p;  p += M * HID * 2;
  __bf16* Wq_bf  = (__bf16*)p;  p += (size_t)HID * HID * 2;
  __bf16* Wk_bf  = (__bf16*)p;  p += KVD * HID * 2;
  __bf16* Wv_bf  = (__bf16*)p;  p += KVD * HID * 2;
  __bf16* Wo_bf  = (__bf16*)p;  p += (size_t)HID * HID * 2;
  float*  Qw     = (float*)p;   p += M * HID * 4;
  float*  Kw     = (float*)p;   p += M * KVD * 4;
  float*  Vw     = (float*)p;   p += M * KVD * 4;
  __bf16* Ow_bf  = (__bf16*)p;  p += M * HID * 2;

  hipMemcpyAsync(cache_out, cache_in, (size_t)MAXP * 2 * PAGE_STRIDE * sizeof(float),
                 hipMemcpyDeviceToDevice, stream);

  // one-time fp32 -> bf16 conversion of GEMM operands
  cvt_f32_bf16<<<(int)(M * HID / 1024), 256, 0, stream>>>(hidden, hid_bf, (int)(M * HID / 4));
  cvt_f32_bf16<<<(int)((size_t)HID * HID / 1024), 256, 0, stream>>>(Wq, Wq_bf, (int)((size_t)HID * HID / 4));
  cvt_f32_bf16<<<(int)(KVD * HID / 1024), 256, 0, stream>>>(Wk, Wk_bf, (int)(KVD * HID / 4));
  cvt_f32_bf16<<<(int)(KVD * HID / 1024), 256, 0, stream>>>(Wv, Wv_bf, (int)(KVD * HID / 4));
  cvt_f32_bf16<<<(int)((size_t)HID * HID / 1024), 256, 0, stream>>>(Wo, Wo_bf, (int)((size_t)HID * HID / 4));

  gemm_bf16_async<<<dim3(HID / 128, M / 128), 256, 0, stream>>>(hid_bf, Wq_bf, Qw, M, HID, HID);
  gemm_bf16_async<<<dim3(KVD / 128, M / 128), 256, 0, stream>>>(hid_bf, Wk_bf, Kw, M, KVD, HID);
  gemm_bf16_async<<<dim3(KVD / 128, M / 128), 256, 0, stream>>>(hid_bf, Wv_bf, Vw, M, KVD, HID);

  kv_append_kernel<<<(int)(M * KVD / 256), 256, 0, stream>>>(Kw, Vw, kv_indices, cache_out, idx_out);

  attn_kernel<<<dim3(QLEN / 64, NH, B_), 128, 0, stream>>>(Qw, cache_out, kv_indices, Ow_bf);

  gemm_bf16_async<<<dim3(HID / 128, M / 128), 256, 0, stream>>>(Ow_bf, Wo_bf, out, M, HID, HID);
}
